// MultiHeadAttention_40252433498326
// MI455X (gfx1250) — matrix operand order repaired and checked by reading, not compiled, not run on hardware
//
#include <hip/hip_runtime.h>
#include <hip/hip_bf16.h>

typedef _Float16 h16;
typedef __attribute__((ext_vector_type(16))) _Float16 v16h;
typedef __attribute__((ext_vector_type(8)))  _Float16 v8h;
typedef __attribute__((ext_vector_type(8)))  float    v8f;

#define WMMA_F16(a,b,c) __builtin_amdgcn_wmma_f32_16x16x32_f16(false,(a),false,(b),(short)0,(c),false,false)

constexpr int BB = 2, S = 2048, D = 1024, H = 16, HD = 64;
constexpr int M  = BB * S;  // 4096 rows of x / ctx

// ---------------------------------------------------------------------------
// Operand loaders (wave32); A and B share one contiguous k order, k = 16*(lane>>4) + element.
// A (16x32 f16, MxK): lane<16 -> row M=lane, K base 0; lane>=16 -> row M=lane-16, K base 16.
//   halves 0..15 = K[base+0 .. base+15].
// B (32x16 f16, KxN): lane holds column N=lane&15; 16 contiguous K starting at
//   (lane<16 ? 0 : 16).
// C/D (16x16 f32): VGPR r -> M = (lane<16 ? r : r+8), N = lane&15.
// ---------------------------------------------------------------------------
__device__ __forceinline__ v16h load_a_frag(const h16* __restrict__ A, int lda,
                                            int row0, int k0, int lane) {
  int r  = row0 + (lane & 15);
  int kk = k0 + ((lane >> 4) << 4);
  const h16* p = A + (size_t)r * lda + kk;
  v8h lo = *(const v8h*)p;
  v8h hi = *(const v8h*)(p + 8);
  v16h out;
#pragma unroll
  for (int i = 0; i < 8; ++i) { out[i] = lo[i]; out[i + 8] = hi[i]; }
  return out;
}

__device__ __forceinline__ v16h load_b_frag(const h16* __restrict__ Bt, int ldb,
                                            int n0, int k0, int lane) {
  int c  = n0 + (lane & 15);
  int kk = k0 + ((lane >> 4) << 4);
  return *(const v16h*)(Bt + (size_t)c * ldb + kk);
}

__device__ __forceinline__ v16h load_b_frag_clamp(const h16* __restrict__ Bt, int ldb,
                                                  int n0, int k0, int lane, int maxcol) {
  int c = n0 + (lane & 15);
  if (c > maxcol) c = maxcol;              // masked columns -> value irrelevant
  int kk = k0 + ((lane >> 4) << 4);
  return *(const v16h*)(Bt + (size_t)c * ldb + kk);
}

// ---------------------------------------------------------------------------
// fp32 -> f16 elementwise convert (grid-stride)
// ---------------------------------------------------------------------------
__global__ __launch_bounds__(256) void cvt_f32_f16(const float* __restrict__ in,
                                                   h16* __restrict__ out, int n) {
  for (int i = blockIdx.x * blockDim.x + threadIdx.x; i < n; i += gridDim.x * blockDim.x)
    out[i] = (h16)in[i];
}

// ---------------------------------------------------------------------------
// Transpose+convert DxD fp32 weight -> f16  Wt[n][k] = W[k][n]
// ---------------------------------------------------------------------------
__global__ __launch_bounds__(256) void transpose_cvt(const float* __restrict__ W,
                                                     h16* __restrict__ Wt) {
  __shared__ h16 t[32][33];
  int bx = blockIdx.x * 32;     // n tile
  int by = blockIdx.y * 32;     // k tile
  int lx = threadIdx.x & 31;
  int ly = threadIdx.x >> 5;    // 8 rows per pass
#pragma unroll
  for (int i = 0; i < 32; i += 8)
    t[ly + i][lx] = (h16)W[(size_t)(by + ly + i) * D + bx + lx];
  __syncthreads();
#pragma unroll
  for (int i = 0; i < 32; i += 8)
    Wt[(size_t)(bx + ly + i) * D + by + lx] = t[lx][ly + i];
}

// ---------------------------------------------------------------------------
// WMMA GEMM:  Y(MxD) = A(MxD,f16 row-major) * Bt(DxD,f16 N-major)
// Block = 128x128 tile, 8 waves, each wave 32x64 (2x4 accumulators).
// mode 0/1: write f16 (B,H,S,HD)   (Q / K)
// mode 2  : write f16 (B,H,HD,S)   (V transposed)
// mode 3  : write fp32 row-major   (final output)
// A and B tiles are read straight from global: weights (2 MB) and x (8 MB)
// live in the 192 MB L2, so LDS staging buys nothing here.
// ---------------------------------------------------------------------------
__global__ __launch_bounds__(256) void gemm_wmma(const h16* __restrict__ A,
                                                 const h16* __restrict__ Bt,
                                                 h16* __restrict__ oh,
                                                 float* __restrict__ of,
                                                 int mode) {
  int wave = threadIdx.x >> 5;
  int lane = threadIdx.x & 31;
  int wm = blockIdx.x * 128 + (wave & 3) * 32;   // 2 m-subtiles of 16
  int wn = blockIdx.y * 128 + (wave >> 2) * 64;  // 4 n-subtiles of 16

  v8f acc[2][4] = {};
  for (int k0 = 0; k0 < D; k0 += 32) {
    v16h a0 = load_a_frag(A, D, wm,      k0, lane);
    v16h a1 = load_a_frag(A, D, wm + 16, k0, lane);
#pragma unroll
    for (int j = 0; j < 4; ++j) {
      v16h b = load_b_frag(Bt, D, wn + j * 16, k0, lane);
      acc[0][j] = WMMA_F16(a0, b, acc[0][j]);
      acc[1][j] = WMMA_F16(a1, b, acc[1][j]);
    }
  }

  int lm = (lane >> 4) << 3;  // +0 or +8 row group
#pragma unroll
  for (int i = 0; i < 2; ++i) {
#pragma unroll
    for (int j = 0; j < 4; ++j) {
      int n = wn + j * 16 + (lane & 15);
#pragma unroll
      for (int r = 0; r < 8; ++r) {
        int mrow = wm + i * 16 + lm + r;
        float v = acc[i][j][r];
        if (mode == 3) {
          of[(size_t)mrow * D + n] = v;
        } else {
          int b  = mrow >> 11;        // / S
          int s  = mrow & (S - 1);
          int h  = n >> 6;            // / HD
          int hd = n & (HD - 1);
          if (mode == 2)
            oh[(((size_t)(b * H + h)) * HD + hd) * S + s] = (h16)v;
          else
            oh[(((size_t)(b * H + h)) * S + s) * HD + hd] = (h16)v;
        }
      }
    }
  }
}

// ---------------------------------------------------------------------------
// Causal flash attention. grid = (B*H, S/128). 8 waves/block, one 16-row
// q-tile per wave, streaming 32 k-columns per step with online softmax.
// No block-wide barriers (waves have different causal trip counts);
// the P tile C->A relayout bounces through wave-private LDS.
// ---------------------------------------------------------------------------
__global__ __launch_bounds__(256) void attn_wmma(const h16* __restrict__ Q,
                                                 const h16* __restrict__ K,
                                                 const h16* __restrict__ Vt,
                                                 h16* __restrict__ ctx) {
  __shared__ h16 pbuf[8][16 * 32];

  int bh   = blockIdx.x;               // b*H + h
  int b    = bh >> 4;
  int h    = bh & (H - 1);
  int wave = threadIdx.x >> 5;
  int lane = threadIdx.x & 31;
  int q0   = blockIdx.y * 128 + wave * 16;

  const h16* Qb = Q  + (size_t)bh * S * HD;
  const h16* Kb = K  + (size_t)bh * S * HD;
  const h16* Vb = Vt + (size_t)bh * HD * S;
  h16* pb = pbuf[wave];

  // Q tile as two A-operand fragments (HD = 64 = 2 x k32)
  v16h qf0 = load_a_frag(Qb, HD, q0, 0,  lane);
  v16h qf1 = load_a_frag(Qb, HD, q0, 32, lane);

  v8f o[4] = {};                       // 16x64 output accumulator (4 C tiles)
  float mrow[8], lrow[8];
#pragma unroll
  for (int r = 0; r < 8; ++r) { mrow[r] = -1e30f; lrow[r] = 0.f; }

  int rowbase = q0 + ((lane >> 4) << 3);  // row of acc VGPR r = rowbase + r
  int colA    = lane & 15;
  const float sc = 0.125f;                // 1/sqrt(64)

  for (int k0 = 0; k0 <= q0 + 15; k0 += 32) {
    // ---- scores: two 16x16 tiles, k-dim = HD (2 chained WMMAs each) ----
    v8f s0 = {}, s1 = {};
    {
      v16h b00 = load_b_frag_clamp(Kb, HD, k0,      0,  lane, S - 1);
      v16h b01 = load_b_frag_clamp(Kb, HD, k0,      32, lane, S - 1);
      s0 = WMMA_F16(qf0, b00, s0);
      s0 = WMMA_F16(qf1, b01, s0);
      v16h b10 = load_b_frag_clamp(Kb, HD, k0 + 16, 0,  lane, S - 1);
      v16h b11 = load_b_frag_clamp(Kb, HD, k0 + 16, 32, lane, S - 1);
      s1 = WMMA_F16(qf0, b10, s1);
      s1 = WMMA_F16(qf1, b11, s1);
    }

    // ---- scale + causal mask + row max ----
    float cmax[8];
#pragma unroll
    for (int r = 0; r < 8; ++r) {
      int row = rowbase + r;
      float x0 = s0[r] * sc; if (k0 + colA      > row) x0 = -1e30f;
      float x1 = s1[r] * sc; if (k0 + 16 + colA > row) x1 = -1e30f;
      s0[r] = x0; s1[r] = x1;
      cmax[r] = fmaxf(x0, x1);
    }
#pragma unroll
    for (int off = 1; off < 16; off <<= 1)
#pragma unroll
      for (int r = 0; r < 8; ++r)
        cmax[r] = fmaxf(cmax[r], __shfl_xor(cmax[r], off, 32));

    // ---- online softmax update ----
    float alpha[8], rsum[8];
#pragma unroll
    for (int r = 0; r < 8; ++r) {
      float mn = fmaxf(mrow[r], cmax[r]);
      alpha[r] = __expf(mrow[r] - mn);
      mrow[r]  = mn;
      float p0 = __expf(s0[r] - mn);
      float p1 = __expf(s1[r] - mn);
      s0[r] = p0; s1[r] = p1;
      rsum[r] = p0 + p1;
    }
#pragma unroll
    for (int off = 1; off < 16; off <<= 1)
#pragma unroll
      for (int r = 0; r < 8; ++r)
        rsum[r] += __shfl_xor(rsum[r], off, 32);
#pragma unroll
    for (int r = 0; r < 8; ++r) lrow[r] = lrow[r] * alpha[r] + rsum[r];
#pragma unroll
    for (int t = 0; t < 4; ++t)
#pragma unroll
      for (int r = 0; r < 8; ++r) o[t][r] *= alpha[r];

    // ---- relayout P (C layout) -> A operand via wave-private LDS ----
    int prow = (lane >> 4) << 3;
#pragma unroll
    for (int r = 0; r < 8; ++r) {
      pb[(prow + r) * 32 + colA]      = (h16)s0[r];
      pb[(prow + r) * 32 + colA + 16] = (h16)s1[r];
    }
    asm volatile("s_wait_dscnt 0" ::: "memory");
    v16h pf = load_a_frag(pb, 32, 0, 0, lane);

    // ---- o += P(16x32) * V(32x64) ----
#pragma unroll
    for (int t = 0; t < 4; ++t) {
      int hd = t * 16 + (lane & 15);
      int kk = k0 + ((lane >> 4) << 4);
      if (kk > S - 16) kk = S - 16;     // masked rows have p == 0
      v16h vf = *(const v16h*)(Vb + (size_t)hd * S + kk);
      o[t] = WMMA_F16(pf, vf, o[t]);
    }
  }

  // ---- normalize and write context as f16 (B,S,D) ----
#pragma unroll
  for (int r = 0; r < 8; ++r) {
    float inv = 1.0f / lrow[r];
    int srow = rowbase + r;
    size_t base = ((size_t)(b * S + srow)) * D + h * HD;
#pragma unroll
    for (int t = 0; t < 4; ++t)
      ctx[base + t * 16 + (lane & 15)] = (h16)(o[t][r] * inv);
  }
}

// ---------------------------------------------------------------------------
extern "C" void kernel_launch(void* const* d_in, const int* in_sizes, int n_in,
                              void* d_out, int out_size, void* d_ws, size_t ws_size,
                              hipStream_t stream) {
  const float* x  = (const float*)d_in[0];
  const float* WQ = (const float*)d_in[1];
  const float* WK = (const float*)d_in[2];
  const float* WV = (const float*)d_in[3];
  const float* WO = (const float*)d_in[4];
  // d_in[5] = causal mask: known tril, applied analytically in attn_wmma.
  float* out = (float*)d_out;

  char* ws = (char*)d_ws;
  const size_t MB = 1024 * 1024;
  h16* x16  = (h16*)(ws + 0 * MB);   // 8 MB
  h16* wqt  = (h16*)(ws + 8 * MB);   // 2 MB each, transposed f16
  h16* wkt  = (h16*)(ws + 10 * MB);
  h16* wvt  = (h16*)(ws + 12 * MB);
  h16* wot  = (h16*)(ws + 14 * MB);
  h16* q16  = (h16*)(ws + 16 * MB);  // (B,H,S,HD) 8 MB
  h16* k16  = (h16*)(ws + 24 * MB);  // (B,H,S,HD) 8 MB
  h16* vT16 = (h16*)(ws + 32 * MB);  // (B,H,HD,S) 8 MB
  h16* ctx  = (h16*)(ws + 40 * MB);  // (B,S,D)    8 MB

  cvt_f32_f16<<<2048, 256, 0, stream>>>(x, x16, M * D);
  transpose_cvt<<<dim3(32, 32), 256, 0, stream>>>(WQ, wqt);
  transpose_cvt<<<dim3(32, 32), 256, 0, stream>>>(WK, wkt);
  transpose_cvt<<<dim3(32, 32), 256, 0, stream>>>(WV, wvt);
  transpose_cvt<<<dim3(32, 32), 256, 0, stream>>>(WO, wot);

  gemm_wmma<<<dim3(M / 128, D / 128), 256, 0, stream>>>(x16, wqt, q16,  nullptr, 0);
  gemm_wmma<<<dim3(M / 128, D / 128), 256, 0, stream>>>(x16, wkt, k16,  nullptr, 1);
  gemm_wmma<<<dim3(M / 128, D / 128), 256, 0, stream>>>(x16, wvt, vT16, nullptr, 2);

  attn_wmma<<<dim3(BB * H, S / 128), 256, 0, stream>>>(q16, k16, vT16, ctx);

  gemm_wmma<<<dim3(M / 128, D / 128), 256, 0, stream>>>(ctx, wot, nullptr, out, 3);
}
